// Encode_65249143160985
// MI455X (gfx1250) — matrix operand order repaired and checked by reading, not compiled, not run on hardware
//
#include <hip/hip_runtime.h>
#include <hip/hip_bf16.h>

typedef __attribute__((ext_vector_type(16))) _Float16 v16h;
typedef __attribute__((ext_vector_type(8)))  _Float16 v8h;
typedef __attribute__((ext_vector_type(8)))  float    v8f;

#define N_NODES  200000
#define N_EDGES  3200000
#define N_GRAPHS 64

#define SELU_L  1.0507009873554805f
#define SELU_LA 1.7580993408473766f   // SELU_L * SELU_A

// Branchless SELU: exact for both signs, no EXEC-mask divergence.
//  x > 0: max=x, min=0 -> L*x + LA*(e^0-1) = L*x
//  x <= 0: max=0, min=x -> LA*(e^x - 1)
__device__ __forceinline__ float selu(float x) {
    float p = fmaxf(x, 0.0f);
    float m = fminf(x, 0.0f);
    return SELU_L * p + SELU_LA * (__expf(m) - 1.0f);
}

// Load a 32x16 f16 B-fragment (K rows x 16 cols) from an f32 weight matrix
// W[K_in][N_out] (row-major), taking columns [coloff, coloff+16). Out-of-range
// rows/cols are zero so K/N padding in A / C is a don't-care.
__device__ __forceinline__ v16h load_bfrag(const float* __restrict__ W,
                                           int K_in, int N_out, int coloff,
                                           int n, int g) {
    v16h b;
    int col = coloff + n;
    int cc = col < N_out ? col : 0;
#pragma unroll
    for (int i = 0; i < 16; ++i) {
        int k  = (i < 8) ? (g * 8 + i) : (16 + g * 8 + (i - 8));
        int kk = k < K_in ? k : 0;
        float v = W[kk * N_out + cc];
        b[i] = (k < K_in && col < N_out) ? (_Float16)v : (_Float16)0.0f;
    }
    return b;
}

__device__ __forceinline__ v8f bias_frag(float b) {
    v8f c;
#pragma unroll
    for (int i = 0; i < 8; ++i) c[i] = b;
    return c;
}

__device__ __forceinline__ v16h lds_afrag(const _Float16* st, int r, int g) {
    // A elements 0..7  = K [g*8 .. g*8+7]       -> st[r*32 + g*8 ..]
    // A elements 8..15 = K [16+g*8 .. 16+g*8+7] -> st[r*32 + 16 + g*8 ..]
    const v8h lo = *(const v8h*)(st + r * 32 + g * 8);
    const v8h hi = *(const v8h*)(st + r * 32 + 16 + g * 8);
    v16h a;
#pragma unroll
    for (int i = 0; i < 8; ++i) { a[i] = lo[i]; a[8 + i] = hi[i]; }
    return a;
}

// ---------------- Edge block: [E,31] -> 25 -> 20 -> 10 (WMMA) ---------------
__global__ void __launch_bounds__(256) edge_kernel(
    const float* __restrict__ nodes, const float* __restrict__ edges,
    const float* __restrict__ ew1, const float* __restrict__ eb1,
    const float* __restrict__ ew2, const float* __restrict__ eb2,
    const float* __restrict__ ew3, const float* __restrict__ eb3,
    const int* __restrict__ senders, const int* __restrict__ receivers,
    const int* __restrict__ edge_graph,
    float* __restrict__ e_out, float* __restrict__ agg_e2n,
    float* __restrict__ agg_e2g)
{
    __shared__ __align__(16) _Float16 stage[8][512]; // 16x32 halfs per wave
    __shared__ float gacc[N_GRAPHS * 10];
    for (int i = threadIdx.x; i < N_GRAPHS * 10; i += blockDim.x) gacc[i] = 0.0f;
    __syncthreads();

    const int lane = threadIdx.x & 31;
    const int wave = threadIdx.x >> 5;
    const int n = lane & 15;   // column (B/C/D) or row (A)
    const int g = lane >> 4;   // half-wave group
    _Float16* st = stage[wave];

    // Loop-invariant B fragments; biases pre-loaded into C fragments.
    const v16h B1a = load_bfrag(ew1, 31, 25,  0, n, g);
    const v16h B1b = load_bfrag(ew1, 31, 25, 16, n, g);
    const v16h B2a = load_bfrag(ew2, 25, 20,  0, n, g);
    const v16h B2b = load_bfrag(ew2, 25, 20, 16, n, g);
    const v16h B3  = load_bfrag(ew3, 20, 10,  0, n, g);
    const v8f Cb1a = bias_frag(eb1[n]);
    const v8f Cb1b = bias_frag((n < 9)  ? eb1[16 + n] : 0.0f);
    const v8f Cb2a = bias_frag(eb2[n]);
    const v8f Cb2b = bias_frag((n < 4)  ? eb2[16 + n] : 0.0f);
    const v8f Cb3  = bias_frag((n < 10) ? eb3[n]      : 0.0f);

    const int nTiles = N_EDGES / 16;           // 200000, exact
    const int stride = gridDim.x * 8;
    for (int t = blockIdx.x * 8 + wave; t < nTiles; t += stride) {
        const int base = t * 16;
        // ---- assemble A1: 16 edges x 32 features (31 real + 1 pad) ----
        const int e   = base + n;              // lanes L and L+16 share row n
        const int rcv = receivers[e];
        const int snd = senders[e];
        const float* ep = edges + (size_t)e   * 5;
        const float* rp = nodes + (size_t)rcv * 13;
        const float* sp = nodes + (size_t)snd * 13;
        v16h A1;
#pragma unroll
        for (int i = 0; i < 16; ++i) {
            int k  = (i < 8) ? (g * 8 + i) : (16 + g * 8 + (i - 8));
            int kk = k > 30 ? 30 : k;          // k==31 is padding (B row is 0)
            const float* p = (kk < 5) ? (ep + kk)
                           : (kk < 18) ? (rp + kk - 5)
                                       : (sp + kk - 18);
            A1[i] = (_Float16)(*p);
        }
        // ---- layer 1: 31 -> 25 (bias via C operand) ----
        v8f c0 = __builtin_amdgcn_wmma_f32_16x16x32_f16(false, A1, false, B1a, (short)0, Cb1a, false, false);
        v8f c1 = __builtin_amdgcn_wmma_f32_16x16x32_f16(false, A1, false, B1b, (short)0, Cb1b, false, false);
#pragma unroll
        for (int i = 0; i < 8; ++i) {
            int m = g * 8 + i;
            st[m * 32 + n]      = (_Float16)selu(c0[i]);
            st[m * 32 + 16 + n] = (_Float16)selu(c1[i]);
        }
        asm volatile("s_wait_dscnt 0" ::: "memory");
        const v16h A2 = lds_afrag(st, n, g);
        // ---- layer 2: 25 -> 20 ----
        v8f d0 = __builtin_amdgcn_wmma_f32_16x16x32_f16(false, A2, false, B2a, (short)0, Cb2a, false, false);
        v8f d1 = __builtin_amdgcn_wmma_f32_16x16x32_f16(false, A2, false, B2b, (short)0, Cb2b, false, false);
#pragma unroll
        for (int i = 0; i < 8; ++i) {
            int m = g * 8 + i;
            st[m * 32 + n]      = (_Float16)selu(d0[i]);
            st[m * 32 + 16 + n] = (_Float16)selu(d1[i]);
        }
        asm volatile("s_wait_dscnt 0" ::: "memory");
        const v16h A3 = lds_afrag(st, n, g);
        // ---- layer 3: 20 -> 10 (no activation) ----
        v8f d3 = __builtin_amdgcn_wmma_f32_16x16x32_f16(false, A3, false, B3, (short)0, Cb3, false, false);
        if (n < 10) {
#pragma unroll
            for (int i = 0; i < 8; ++i) {
                int m  = g * 8 + i;
                int ee = base + m;
                float val = d3[i];
                e_out[(size_t)ee * 10 + n] = val;
                atomicAdd(&agg_e2n[(size_t)receivers[ee] * 10 + n], val);
                atomicAdd(&gacc[edge_graph[ee] * 10 + n], val);
            }
        }
    }
    __syncthreads();
    for (int i = threadIdx.x; i < N_GRAPHS * 10; i += blockDim.x) {
        float v = gacc[i];
        if (v != 0.0f) atomicAdd(&agg_e2g[i], v);
    }
}

// ---------------- Node block: [N,23] -> 18 -> 10 (WMMA) ---------------------
__global__ void __launch_bounds__(256) node_kernel(
    const float* __restrict__ nodes, const float* __restrict__ agg_e2n,
    const float* __restrict__ nw1, const float* __restrict__ nb1,
    const float* __restrict__ nw2, const float* __restrict__ nb2,
    const int* __restrict__ node_graph,
    float* __restrict__ n_out, float* __restrict__ agg_n2g)
{
    __shared__ __align__(16) _Float16 stage[8][512];
    __shared__ float gacc[N_GRAPHS * 10];
    for (int i = threadIdx.x; i < N_GRAPHS * 10; i += blockDim.x) gacc[i] = 0.0f;
    __syncthreads();

    const int lane = threadIdx.x & 31;
    const int wave = threadIdx.x >> 5;
    const int n = lane & 15;
    const int g = lane >> 4;
    _Float16* st = stage[wave];

    const v16h B1a = load_bfrag(nw1, 23, 18,  0, n, g);
    const v16h B1b = load_bfrag(nw1, 23, 18, 16, n, g);
    const v16h B2  = load_bfrag(nw2, 18, 10,  0, n, g);
    const v8f Cb1a = bias_frag(nb1[n]);
    const v8f Cb1b = bias_frag((n < 2)  ? nb1[16 + n] : 0.0f);
    const v8f Cb2  = bias_frag((n < 10) ? nb2[n]      : 0.0f);

    const int nTiles = N_NODES / 16;           // 12500, exact
    const int stride = gridDim.x * 8;
    for (int t = blockIdx.x * 8 + wave; t < nTiles; t += stride) {
        const int base = t * 16;
        const int nd = base + n;
        const float* ap = agg_e2n + (size_t)nd * 10;
        const float* np = nodes   + (size_t)nd * 13;
        v16h A1;
#pragma unroll
        for (int i = 0; i < 16; ++i) {
            int k  = (i < 8) ? (g * 8 + i) : (16 + g * 8 + (i - 8));
            int kk = k > 22 ? 22 : k;          // K pad (B rows zero)
            const float* p = (kk < 10) ? (ap + kk) : (np + kk - 10);
            A1[i] = (_Float16)(*p);
        }
        v8f c0 = __builtin_amdgcn_wmma_f32_16x16x32_f16(false, A1, false, B1a, (short)0, Cb1a, false, false);
        v8f c1 = __builtin_amdgcn_wmma_f32_16x16x32_f16(false, A1, false, B1b, (short)0, Cb1b, false, false);
#pragma unroll
        for (int i = 0; i < 8; ++i) {
            int m = g * 8 + i;
            st[m * 32 + n]      = (_Float16)selu(c0[i]);
            st[m * 32 + 16 + n] = (_Float16)selu(c1[i]);
        }
        asm volatile("s_wait_dscnt 0" ::: "memory");
        const v16h A2 = lds_afrag(st, n, g);
        v8f d = __builtin_amdgcn_wmma_f32_16x16x32_f16(false, A2, false, B2, (short)0, Cb2, false, false);
        if (n < 10) {
#pragma unroll
            for (int i = 0; i < 8; ++i) {
                int m  = g * 8 + i;
                int nn = base + m;
                float val = d[i];
                n_out[(size_t)nn * 10 + n] = val;
                atomicAdd(&gacc[node_graph[nn] * 10 + n], val);
            }
        }
    }
    __syncthreads();
    for (int i = threadIdx.x; i < N_GRAPHS * 10; i += blockDim.x) {
        float v = gacc[i];
        if (v != 0.0f) atomicAdd(&agg_n2g[i], v);
    }
}

// ---------------- Global block: [G,20] -> 15 -> 15 -> 10 (scalar) -----------
__global__ void __launch_bounds__(64) global_kernel(
    const float* __restrict__ agg_e2g, const float* __restrict__ agg_n2g,
    const float* __restrict__ gw1, const float* __restrict__ gb1,
    const float* __restrict__ gw2, const float* __restrict__ gb2,
    const float* __restrict__ gw3, const float* __restrict__ gb3,
    float* __restrict__ g_out)
{
    int gi = threadIdx.x;
    if (gi >= N_GRAPHS) return;
    float gin[20];
#pragma unroll
    for (int k = 0; k < 10; ++k) {
        gin[k]      = agg_e2g[gi * 10 + k];
        gin[10 + k] = agg_n2g[gi * 10 + k];
    }
    float h1[15];
#pragma unroll
    for (int j = 0; j < 15; ++j) {
        float s = gb1[j];
        for (int k = 0; k < 20; ++k) s += gin[k] * gw1[k * 15 + j];
        h1[j] = selu(s);
    }
    float h2[15];
#pragma unroll
    for (int j = 0; j < 15; ++j) {
        float s = gb2[j];
        for (int k = 0; k < 15; ++k) s += h1[k] * gw2[k * 15 + j];
        h2[j] = selu(s);
    }
#pragma unroll
    for (int j = 0; j < 10; ++j) {
        float s = gb3[j];
        for (int k = 0; k < 15; ++k) s += h2[k] * gw3[k * 10 + j];
        g_out[gi * 10 + j] = s;
    }
}

extern "C" void kernel_launch(void* const* d_in, const int* in_sizes, int n_in,
                              void* d_out, int out_size, void* d_ws, size_t ws_size,
                              hipStream_t stream) {
    const float* nodes = (const float*)d_in[0];
    const float* edges = (const float*)d_in[1];
    const float* ew1 = (const float*)d_in[2];  const float* eb1 = (const float*)d_in[3];
    const float* ew2 = (const float*)d_in[4];  const float* eb2 = (const float*)d_in[5];
    const float* ew3 = (const float*)d_in[6];  const float* eb3 = (const float*)d_in[7];
    const float* nw1 = (const float*)d_in[8];  const float* nb1 = (const float*)d_in[9];
    const float* nw2 = (const float*)d_in[10]; const float* nb2 = (const float*)d_in[11];
    const float* gw1 = (const float*)d_in[12]; const float* gb1 = (const float*)d_in[13];
    const float* gw2 = (const float*)d_in[14]; const float* gb2 = (const float*)d_in[15];
    const float* gw3 = (const float*)d_in[16]; const float* gb3 = (const float*)d_in[17];
    const int* senders    = (const int*)d_in[18];
    const int* receivers  = (const int*)d_in[19];
    const int* node_graph = (const int*)d_in[20];
    const int* edge_graph = (const int*)d_in[21];

    float* e_out = (float*)d_out;
    float* n_out = e_out + (size_t)N_EDGES * 10;
    float* g_out = n_out + (size_t)N_NODES * 10;

    float* agg_e2n = (float*)d_ws;                       // N_NODES*10 f32
    float* agg_e2g = agg_e2n + (size_t)N_NODES * 10;     // 640 f32
    float* agg_n2g = agg_e2g + N_GRAPHS * 10;            // 640 f32

    hipMemsetAsync(d_ws, 0,
                   ((size_t)N_NODES * 10 + 2 * N_GRAPHS * 10) * sizeof(float),
                   stream);

    edge_kernel<<<1024, 256, 0, stream>>>(nodes, edges, ew1, eb1, ew2, eb2,
                                          ew3, eb3, senders, receivers,
                                          edge_graph, e_out, agg_e2n, agg_e2g);
    node_kernel<<<512, 256, 0, stream>>>(nodes, agg_e2n, nw1, nb1, nw2, nb2,
                                         node_graph, n_out, agg_n2g);
    global_kernel<<<1, 64, 0, stream>>>(agg_e2g, agg_n2g, gw1, gb1, gw2, gb2,
                                        gw3, gb3, g_out);
}
